// GAT_7241314861277
// MI455X (gfx1250) — compile-verified, matrix operand order checked by reading, not run on hardware
//
#include <hip/hip_runtime.h>
#include <hip/hip_bf16.h>

#define NN     20000
#define NE     160000
#define EPLUS  (NE + NN)
#define FF     256
#define HH     8
#define DD     32
#define OUTF   128
#define LM1    5
#define EPSV   1e-5f

typedef __attribute__((ext_vector_type(16))) __bf16 v16bf;
typedef __attribute__((ext_vector_type(8)))  float  v8f;
typedef __attribute__((ext_vector_type(4)))  int    v4i;

struct __align__(16) U4 { unsigned x, y, z, w; };
union FragBF { v16bf v; U4 q[2]; unsigned short u[16]; };

// CDNA5 async global->LDS copy (ASYNCcnt), with compile-safe fallback.
// Signature (from toolchain): (v4i AS(1)* src, v4i AS(3)* dst, Ii offset, Ii cpol)
#if __has_builtin(__builtin_amdgcn_global_load_async_to_lds_b128)
#define HAVE_ASYNC_LDS 1
#define ASYNC_LDS_COPY_B128(gptr, lptr)                                        \
  __builtin_amdgcn_global_load_async_to_lds_b128(                              \
      (__attribute__((address_space(1))) v4i*)(gptr),                          \
      (__attribute__((address_space(3))) v4i*)(lptr), 0, 0)
#endif

__device__ __forceinline__ unsigned short f32_to_bf16_rn(float f) {
  unsigned u = __float_as_uint(f);
  u += 0x7FFFu + ((u >> 16) & 1u);
  return (unsigned short)(u >> 16);
}

// Monotonic float -> uint encoding (for atomicMax-based segment max).
__device__ __forceinline__ unsigned encF(float f) {
  unsigned u = __float_as_uint(f);
  return (u & 0x80000000u) ? ~u : (u | 0x80000000u);
}
__device__ __forceinline__ float decF(unsigned u) {
  return (u & 0x80000000u) ? __uint_as_float(u ^ 0x80000000u) : __uint_as_float(~u);
}

// ---------------------------------------------------------------- utilities
__global__ void k_zero_f32(float* __restrict__ p, long n) {
  long i = (long)blockIdx.x * blockDim.x + threadIdx.x;
  for (; i < n; i += (long)gridDim.x * blockDim.x) p[i] = 0.0f;
}

// f32 -> packed bf16x2 (n2 = element pairs)
__global__ void k_cvt_bf16(const float* __restrict__ src, unsigned* __restrict__ dst, long n2) {
  long i = (long)blockIdx.x * blockDim.x + threadIdx.x;
  if (i >= n2) return;
  float a = src[2 * i], b = src[2 * i + 1];
  dst[i] = (unsigned)f32_to_bf16_rn(a) | ((unsigned)f32_to_bf16_rn(b) << 16);
}

// f32 [K][Ncols] row-major -> bf16 [Ncols][K] (column-major weights)
__global__ void k_cvtT_bf16(const float* __restrict__ src, unsigned short* __restrict__ dst,
                            int K, int Ncols) {
  int i = blockIdx.x * blockDim.x + threadIdx.x;
  if (i >= K * Ncols) return;
  int r = i / Ncols, c = i - r * Ncols;
  dst[(size_t)c * K + r] = f32_to_bf16_rn(src[i]);
}

// ---------------------------------------------------------------- GCN stage
__global__ void k_edge_deg(const int* __restrict__ ei, float* __restrict__ deg) {
  int i = blockIdx.x * blockDim.x + threadIdx.x;
  if (i >= EPLUS) return;
  int d = (i < NE) ? ei[NE + i] : (i - NE);
  atomicAdd(&deg[d], 1.0f);
}

__global__ void k_dinv(const float* __restrict__ deg, float* __restrict__ dinv) {
  int i = blockIdx.x * blockDim.x + threadIdx.x;
  if (i >= NN) return;
  float d = deg[i];
  dinv[i] = (d > 0.0f) ? rsqrtf(d) : 0.0f;
}

__global__ void k_gcn_scatter(const int* __restrict__ ei, const float* __restrict__ x,
                              const float* __restrict__ dinv, float* __restrict__ acc) {
  int i = blockIdx.x * blockDim.x + threadIdx.x;
  if (i >= EPLUS) return;
  int s = (i < NE) ? ei[i]      : (i - NE);
  int d = (i < NE) ? ei[NE + i] : (i - NE);
  atomicAdd(&acc[d], x[s] * dinv[s] * dinv[d]);
}

__global__ void k_h0(const float* __restrict__ acc, const float* __restrict__ W0,
                     const float* __restrict__ b0, float* __restrict__ h) {
  long i = (long)blockIdx.x * blockDim.x + threadIdx.x;
  if (i >= (long)NN * FF) return;
  int n = (int)(i >> 8), f = (int)(i & 255);
  float v = acc[n] * W0[f] + b0[f];
  h[i] = fmaxf(v, 0.0f);
}

// --------------------------------------------------- WMMA bf16 GEMM, K=256
// C[M x Ncols] = A[M x 256] * B[256 x Ncols] (+ bias).
// A: bf16 row-major [M][256]. Bt: bf16 column-major [Ncols][256].
// Block: 256 thr = 8 wave32 in a 2(M) x 4(N) grid of 16x16 WMMA tiles
// -> 32x64 block tile. M % 32 == 0, Ncols % 64 == 0, K == 256 (no guards).
// Double-buffered LDS; tile k+1 staged (async global->LDS when available)
// while tile k runs on the WMMA pipes. LDS row stride = 40 ushorts = 80 B
// (16B multiple) so every per-lane fragment is two ds_load_b128's.

__device__ __forceinline__ void gemm_stage(const unsigned short* __restrict__ A,
                                           const unsigned short* __restrict__ Bt,
                                           unsigned short* AshBuf, unsigned short* BshBuf,
                                           int mBlock, int nBlock, int k0, int tid) {
  const int r = tid >> 2, c8 = (tid & 3) << 3;   // A: r<32 (tid<128); B: r<64
#ifdef HAVE_ASYNC_LDS
  if (tid < 128)
    ASYNC_LDS_COPY_B128(&A[(size_t)(mBlock + r) * 256 + k0 + c8], &AshBuf[r * 40 + c8]);
  ASYNC_LDS_COPY_B128(&Bt[(size_t)(nBlock + r) * 256 + k0 + c8], &BshBuf[r * 40 + c8]);
#else
  if (tid < 128)
    *(U4*)&AshBuf[r * 40 + c8] = *(const U4*)&A[(size_t)(mBlock + r) * 256 + k0 + c8];
  *(U4*)&BshBuf[r * 40 + c8] = *(const U4*)&Bt[(size_t)(nBlock + r) * 256 + k0 + c8];
#endif
}

__device__ __forceinline__ void gemm_stage_wait() {
#ifdef HAVE_ASYNC_LDS
#if __has_builtin(__builtin_amdgcn_s_wait_asynccnt)
  __builtin_amdgcn_s_wait_asynccnt(0);
#else
  asm volatile("s_wait_asynccnt 0x0" ::: "memory");
#endif
#endif
  __syncthreads();
}

__global__ void k_gemm_bf16(const unsigned short* __restrict__ A,
                            const unsigned short* __restrict__ Bt,
                            float* __restrict__ C, const float* __restrict__ bias,
                            int M, int Ncols) {
  __shared__ __align__(16) unsigned short Ash[2][32 * 40];
  __shared__ __align__(16) unsigned short Bsh[2][64 * 40];
  const int tid  = threadIdx.x;
  const int lane = tid & 31;
  const int wave = tid >> 5;
  const int mw = wave & 1, nw = wave >> 1;
  const int mBlock = blockIdx.y * 32;
  const int nBlock = blockIdx.x * 64;

  const int khalf = lane >> 4;             // K-half select per WMMA lane layout
  const int kg    = khalf * 8;
  const int arow  = (mw * 16) + (lane & 15);
  const int bcol  = (nw * 16) + (lane & 15);

  gemm_stage(A, Bt, Ash[0], Bsh[0], mBlock, nBlock, 0, tid);
  gemm_stage_wait();

  v8f acc = {};
#pragma unroll
  for (int s = 0; s < 8; ++s) {
    const int cur = s & 1;
    if (s < 7)
      gemm_stage(A, Bt, Ash[cur ^ 1], Bsh[cur ^ 1], mBlock, nBlock, (s + 1) * 32, tid);

    FragBF a, b;
    const unsigned short* Ac = &Ash[cur][arow * 40];
    const unsigned short* Bc = &Bsh[cur][bcol * 40];
    a.q[0] = *(const U4*)&Ac[kg];
    a.q[1] = *(const U4*)&Ac[16 + kg];
    b.q[0] = *(const U4*)&Bc[kg];
    b.q[1] = *(const U4*)&Bc[16 + kg];
    acc = __builtin_amdgcn_wmma_f32_16x16x32_bf16(false, a.v, false, b.v,
                                                  (short)0, acc, false, false);
    if (s < 7) gemm_stage_wait();
  }

  // C/D layout: lane -> column, VGPR v -> row (v + 8*(lane>=16))
  const int col   = nBlock + nw * 16 + (lane & 15);
  const int rbase = mBlock + mw * 16 + (khalf << 3);
  float bv = (bias != nullptr) ? bias[col] : 0.0f;
#pragma unroll
  for (int v2 = 0; v2 < 8; ++v2) {
    C[(size_t)(rbase + v2) * Ncols + col] = acc[v2] + bv;
  }
}

// ----------------------------------------------------- attention dot scores
__global__ void k_att(const float* __restrict__ hW, const float* __restrict__ asrc,
                      const float* __restrict__ adst, float* __restrict__ s,
                      float* __restrict__ t) {
  int n = blockIdx.x;
  int f = threadIdx.x;               // head = f>>5, one head per wave32
  float v  = hW[(long)n * FF + f];
  float ps = v * asrc[f];
  float pt = v * adst[f];
#pragma unroll
  for (int o = 16; o > 0; o >>= 1) {
    ps += __shfl_xor(ps, o, 32);
    pt += __shfl_xor(pt, o, 32);
  }
  if ((f & 31) == 0) {
    int h8 = f >> 5;
    s[n * HH + h8] = ps;
    t[n * HH + h8] = pt;
  }
}

// ------------------------------------------------------ edge softmax passes
__global__ void k_edge_max(const int* __restrict__ ei, const float* __restrict__ s,
                           const float* __restrict__ t, unsigned* __restrict__ mEnc) {
  long i = (long)blockIdx.x * blockDim.x + threadIdx.x;
  if (i >= (long)EPLUS * HH) return;
  int e = (int)(i >> 3), h8 = (int)(i & 7);
  int sn = (e < NE) ? ei[e]      : (e - NE);
  int dn = (e < NE) ? ei[NE + e] : (e - NE);
  float x = s[sn * HH + h8] + t[dn * HH + h8];
  x = (x > 0.0f) ? x : 0.2f * x;                 // leaky_relu(0.2)
  atomicMax(&mEnc[dn * HH + h8], encF(x));
}

__global__ void k_edge_expsum(const int* __restrict__ ei, const float* __restrict__ s,
                              const float* __restrict__ t, const unsigned* __restrict__ mEnc,
                              float* __restrict__ alpha, float* __restrict__ denom) {
  long i = (long)blockIdx.x * blockDim.x + threadIdx.x;
  if (i >= (long)EPLUS * HH) return;
  int e = (int)(i >> 3), h8 = (int)(i & 7);
  int sn = (e < NE) ? ei[e]      : (e - NE);
  int dn = (e < NE) ? ei[NE + e] : (e - NE);
  float x = s[sn * HH + h8] + t[dn * HH + h8];
  x = (x > 0.0f) ? x : 0.2f * x;
  float e2 = __expf(x - decF(mEnc[dn * HH + h8]));
  alpha[i] = e2;
  atomicAdd(&denom[dn * HH + h8], e2);
}

__global__ void k_edge_norm(const int* __restrict__ ei, const float* __restrict__ denom,
                            float* __restrict__ alpha) {
  long i = (long)blockIdx.x * blockDim.x + threadIdx.x;
  if (i >= (long)EPLUS * HH) return;
  int e = (int)(i >> 3), h8 = (int)(i & 7);
  int dn = (e < NE) ? ei[NE + e] : (e - NE);
  alpha[i] /= (denom[dn * HH + h8] + 1e-16f);
}

__global__ void k_edge_scatter(const int* __restrict__ ei, const float* __restrict__ hW,
                               const float* __restrict__ alpha, float* __restrict__ outAcc) {
  long i = (long)blockIdx.x * blockDim.x + threadIdx.x;
  if (i >= (long)EPLUS * FF) return;
  int e = (int)(i >> 8), f = (int)(i & 255);
  int sn = (e < NE) ? ei[e]      : (e - NE);
  int dn = (e < NE) ? ei[NE + e] : (e - NE);
  float a = alpha[e * HH + (f >> 5)];
  atomicAdd(&outAcc[(long)dn * FF + f], a * hW[(long)sn * FF + f]);
}

__global__ void k_residual(float* __restrict__ h, const float* __restrict__ outAcc,
                           const float* __restrict__ bias) {
  long i = (long)blockIdx.x * blockDim.x + threadIdx.x;
  if (i >= (long)NN * FF) return;
  int f = (int)(i & 255);
  float v = outAcc[i] + (bias ? bias[f] : 0.0f);
  h[i] += fmaxf(v, 0.0f);
}

// -------------------------------------------------------------- GraphNorm
__global__ void k_gn_partial(const float* __restrict__ h, float* __restrict__ colSum,
                             float* __restrict__ colSq) {
  int f = threadIdx.x;
  float s = 0.0f, q = 0.0f;
  for (int r = blockIdx.x; r < NN; r += gridDim.x) {
    float v = h[(long)r * FF + f];
    s += v; q += v * v;
  }
  atomicAdd(&colSum[f], s);
  atomicAdd(&colSq[f], q);
}

__global__ void k_gn_finalize(const float* __restrict__ colSum, const float* __restrict__ colSq,
                              const float* __restrict__ gw, const float* __restrict__ gb,
                              const float* __restrict__ gms, float* __restrict__ colScale,
                              float* __restrict__ colShift) {
  int f = threadIdx.x;
  const float invN = 1.0f / (float)NN;
  float mean = colSum[f] * invN;
  float ex2  = colSq[f] * invN;
  float ms   = gms[f];
  // var of (x - ms*mean) = E[x^2] - (2ms - ms^2) * mean^2
  float var = ex2 - (2.0f * ms - ms * ms) * mean * mean;
  float sc  = gw[f] * rsqrtf(var + EPSV);
  colScale[f] = sc;
  colShift[f] = gb[f] - sc * ms * mean;
}

__global__ void k_gn_apply(float* __restrict__ h, const float* __restrict__ colScale,
                           const float* __restrict__ colShift) {
  long i = (long)blockIdx.x * blockDim.x + threadIdx.x;
  if (i >= (long)NN * FF) return;
  int f = (int)(i & 255);
  h[i] = h[i] * colScale[f] + colShift[f];
}

// ------------------------------------------------------------------- launch
extern "C" void kernel_launch(void* const* d_in, const int* in_sizes, int n_in,
                              void* d_out, int out_size, void* d_ws, size_t ws_size,
                              hipStream_t stream) {
  const float* x        = (const float*)d_in[0];
  const float* W0       = (const float*)d_in[1];
  const float* b0       = (const float*)d_in[2];
  const float* gat_lin  = (const float*)d_in[3];   // [5][256][256]
  const float* att_src  = (const float*)d_in[4];   // [5][8][32]
  const float* att_dst  = (const float*)d_in[5];
  const float* gat_bias = (const float*)d_in[6];   // [5][256]
  const float* gn_w     = (const float*)d_in[7];   // [3][256]
  const float* gn_b     = (const float*)d_in[8];
  const float* gn_ms    = (const float*)d_in[9];
  const float* lin_W    = (const float*)d_in[10];  // [256][128]
  const float* lin_b    = (const float*)d_in[11];
  const int*   ei       = (const int*)d_in[12];    // [2][E]
  float*       out      = (float*)d_out;

  // Workspace carve-out (256B aligned slabs).
  char* wsp = (char*)d_ws;
  auto carve = [&](size_t bytes) -> void* {
    void* p = (void*)wsp;
    wsp += (bytes + 255) & ~(size_t)255;
    return p;
  };
  float*          h        = (float*)carve((size_t)NN * FF * 4);
  float*          hW       = (float*)carve((size_t)NN * FF * 4);
  float*          outAcc   = (float*)carve((size_t)NN * FF * 4);
  unsigned short* hbf      = (unsigned short*)carve((size_t)NN * FF * 2);
  unsigned short* WtGat    = (unsigned short*)carve((size_t)LM1 * FF * FF * 2);
  unsigned short* WtFin    = (unsigned short*)carve((size_t)OUTF * FF * 2);
  float*          deg      = (float*)carve((size_t)NN * 4);
  float*          dinv     = (float*)carve((size_t)NN * 4);
  float*          accg     = (float*)carve((size_t)NN * 4);
  float*          sbuf     = (float*)carve((size_t)NN * HH * 4);
  float*          tbuf     = (float*)carve((size_t)NN * HH * 4);
  unsigned*       mEnc     = (unsigned*)carve((size_t)NN * HH * 4);
  float*          denom    = (float*)carve((size_t)NN * HH * 4);
  float*          alphaB   = (float*)carve((size_t)EPLUS * HH * 4);
  float*          colSum   = (float*)carve((size_t)FF * 4);
  float*          colSq    = (float*)carve((size_t)FF * 4);
  float*          colScale = (float*)carve((size_t)FF * 4);
  float*          colShift = (float*)carve((size_t)FF * 4);

  const int T = 256;
  const long NF = (long)NN * FF;
  const long EH = (long)EPLUS * HH;
  const long EF = (long)EPLUS * FF;

  // ---- One-time per launch: convert + transpose weights to bf16 col-major
  for (int i = 0; i < LM1; ++i)
    k_cvtT_bf16<<<dim3((FF * FF + T - 1) / T), T, 0, stream>>>(
        gat_lin + (size_t)i * FF * FF, WtGat + (size_t)i * FF * FF, FF, FF);
  k_cvtT_bf16<<<dim3((FF * OUTF + T - 1) / T), T, 0, stream>>>(lin_W, WtFin, FF, OUTF);

  // ---- GCN input layer (F_in = 1 -> scalar scatter + outer product)
  k_zero_f32<<<dim3((NN + T - 1) / T), T, 0, stream>>>(deg, NN);
  k_zero_f32<<<dim3((NN + T - 1) / T), T, 0, stream>>>(accg, NN);
  k_edge_deg<<<dim3((EPLUS + T - 1) / T), T, 0, stream>>>(ei, deg);
  k_dinv<<<dim3((NN + T - 1) / T), T, 0, stream>>>(deg, dinv);
  k_gcn_scatter<<<dim3((EPLUS + T - 1) / T), T, 0, stream>>>(ei, x, dinv, accg);
  k_h0<<<dim3((unsigned)((NF + T - 1) / T)), T, 0, stream>>>(accg, W0, b0, h);

  // ---- 5 GAT layers
  for (int i = 0; i < LM1; ++i) {
    if (i % 3 == 0) {
      int g = i / 3;
      k_zero_f32<<<dim3(1), T, 0, stream>>>(colSum, FF);
      k_zero_f32<<<dim3(1), T, 0, stream>>>(colSq, FF);
      k_gn_partial<<<dim3(200), T, 0, stream>>>(h, colSum, colSq);
      k_gn_finalize<<<dim3(1), T, 0, stream>>>(colSum, colSq, gn_w + g * FF,
                                               gn_b + g * FF, gn_ms + g * FF,
                                               colScale, colShift);
      k_gn_apply<<<dim3((unsigned)((NF + T - 1) / T)), T, 0, stream>>>(h, colScale, colShift);
    }

    // h -> bf16, then hW = h @ W_i on the WMMA pipes
    k_cvt_bf16<<<dim3((unsigned)((NF / 2 + T - 1) / T)), T, 0, stream>>>(h, (unsigned*)hbf, NF / 2);
    k_gemm_bf16<<<dim3(FF / 64, NN / 32), T, 0, stream>>>(
        hbf, WtGat + (size_t)i * FF * FF, hW, nullptr, NN, FF);

    k_att<<<dim3(NN), T, 0, stream>>>(hW, att_src + (size_t)i * FF,
                                      att_dst + (size_t)i * FF, sbuf, tbuf);

    k_zero_f32<<<dim3((unsigned)(((long)NN * HH + T - 1) / T)), T, 0, stream>>>((float*)mEnc, (long)NN * HH);
    k_zero_f32<<<dim3((unsigned)(((long)NN * HH + T - 1) / T)), T, 0, stream>>>(denom, (long)NN * HH);
    k_zero_f32<<<dim3((unsigned)((NF + T - 1) / T)), T, 0, stream>>>(outAcc, NF);

    unsigned ehBlocks = (unsigned)((EH + T - 1) / T);
    k_edge_max<<<dim3(ehBlocks), T, 0, stream>>>(ei, sbuf, tbuf, mEnc);
    k_edge_expsum<<<dim3(ehBlocks), T, 0, stream>>>(ei, sbuf, tbuf, mEnc, alphaB, denom);
    k_edge_norm<<<dim3(ehBlocks), T, 0, stream>>>(ei, denom, alphaB);
    k_edge_scatter<<<dim3((unsigned)((EF + T - 1) / T)), T, 0, stream>>>(ei, hW, alphaB, outAcc);

    const float* rbias = (i % 3 == 0) ? (gat_bias + (size_t)i * FF) : nullptr;
    k_residual<<<dim3((unsigned)((NF + T - 1) / T)), T, 0, stream>>>(h, outAcc, rbias);
  }

  // ---- final linear (WMMA) with bias, straight into d_out
  k_cvt_bf16<<<dim3((unsigned)((NF / 2 + T - 1) / T)), T, 0, stream>>>(h, (unsigned*)hbf, NF / 2);
  k_gemm_bf16<<<dim3(OUTF / 64, NN / 32), T, 0, stream>>>(hbf, WtFin, out, lin_b, NN, OUTF);
  (void)in_sizes; (void)n_in; (void)out_size; (void)ws_size;
}
